// Hebbian_52707838657193
// MI455X (gfx1250) — matrix operand order repaired and checked by reading, not compiled, not run on hardware
//
#include <hip/hip_runtime.h>
#include <hip/hip_bf16.h>

typedef __attribute__((ext_vector_type(16))) _Float16 v16h;
typedef __attribute__((ext_vector_type(8)))  float    v8f;
typedef __attribute__((ext_vector_type(4)))  float    f32x4;
typedef __attribute__((ext_vector_type(4)))  _Float16 h16x4;

#define DEV static __device__ __forceinline__

// ---------------- LDS layout (bytes), aliased across sync epochs ----------------
// bufA: sKey  -> sA1            bufB: sWkB -> sCorrT
// bufC: sVal  -> sA3            bufD: sWvB -> sRegT
// bufE+bufF: sKencT,sVencT -> a2f (f32)
// bufG: sA(h) -> sWnT           bufH: sB(h) -> sQ(2KB)+sVflat(1KB)
// bufI+bufJ: s(1-w),sW(h) -> Pf (f32)
#define OFF_A      0
#define OFF_B      8192
#define OFF_C      16384
#define OFF_D      24576
#define OFF_E      32768
#define OFF_F      40960
#define OFF_G      49152
#define OFF_H      57344
#define OFF_I      65536
#define OFF_J      73728
#define OFF_W32    81920
#define OFF_SMALL  98304
#define SMEM_BYTES 99328

// A fragment: 16x32 f16, row-major LDS (ld=64). Lane l<16: row=r0+l, K=k0+{0..7,16..23};
// lane>=16: row=r0+l-16, K=k0+{8..15,24..31}.  Two ds_load_b128 per lane.
DEV v16h fragA(const _Float16* M, int r0, int k0, int lane) {
  const int r  = r0 + (lane & 15);
  const int kb = k0 + ((lane >> 4) << 3);
  const _Float16* p = M + r * 64 + kb;
  v16h f;
  _Float16* fp = (_Float16*)&f;
  *(f32x4*)(fp)     = *(const f32x4*)(p);
  *(f32x4*)(fp + 8) = *(const f32x4*)(p + 16);
  return f;
}

// B fragment: 32x16 f16 from N-major LDS (Bt[n*64+k]). Lane l<16: col=c0+l, K=k0+{0..7,16..23};
// lane>=16: col=c0+l-16, K=k0+{8..15,24..31} (matches fragA). Two ds_load_b128 per lane.
DEV v16h fragB(const _Float16* Bt, int c0, int k0, int lane) {
  const int n  = c0 + (lane & 15);
  const int kb = k0 + ((lane >> 4) << 3);
  const _Float16* p = Bt + n * 64 + kb;
  v16h f;
  _Float16* fp = (_Float16*)&f;
  *(f32x4*)(fp)     = *(const f32x4*)(p);
  *(f32x4*)(fp + 8) = *(const f32x4*)(p + 16);
  return f;
}

// One 16x16 output tile of a 64-wide-K GEMM: A row-major, B stored N-major.
DEV v8f mmtile(const _Float16* A, const _Float16* Bt, int ti, int tj, int lane) {
  v8f acc = {0.f, 0.f, 0.f, 0.f, 0.f, 0.f, 0.f, 0.f};
#pragma unroll
  for (int kb = 0; kb < 64; kb += 32) {
    v16h a = fragA(A, ti * 16, kb, lane);
    v16h b = fragB(Bt, tj * 16, kb, lane);
    acc = __builtin_amdgcn_wmma_f32_16x16x32_f16(false, a, false, b, (short)0, acc,
                                                 false, false);
  }
  return acc;
}

// D tile -> f16 N-major (transposed) store; compile-time bias/scale epilogue so no
// runtime branches land between dependent WMMAs. One ds_store_b128 per lane.
template <bool HAS_BIAS, bool HAS_SCALE>
DEV void stN(_Float16* dst, int r0, int c0, v8f acc, int lane,
             const float* bias, const float* rowscale) {
  const int n  = c0 + (lane & 15);
  const int m0 = r0 + ((lane >> 4) << 3);
  const float bb = HAS_BIAS ? bias[n] : 0.f;
  _Float16 tmp[8];
#pragma unroll
  for (int i = 0; i < 8; ++i) {
    float x = acc[i] + bb;
    if (HAS_SCALE) x *= rowscale[m0 + i];
    tmp[i] = (_Float16)x;
  }
  *(f32x4*)(dst + n * 64 + m0) = *(const f32x4*)tmp;
}

DEV void stRh(_Float16* dst, int r0, int c0, v8f acc, int lane) {  // f16 row-major
  const int n  = c0 + (lane & 15);
  const int m0 = r0 + ((lane >> 4) << 3);
#pragma unroll
  for (int i = 0; i < 8; ++i) dst[(m0 + i) * 64 + n] = (_Float16)acc[i];
}

DEV void stRf(float* dst, int r0, int c0, v8f acc, int lane) {      // f32 row-major
  const int n  = c0 + (lane & 15);
  const int m0 = r0 + ((lane >> 4) << 3);
#pragma unroll
  for (int i = 0; i < 8; ++i) dst[(m0 + i) * 64 + n] = acc[i];
}

// Coalesced 64x64 f32 global -> f16 LDS copy (1024 float4 / 256 threads).
// NT=true: single-touch stream, bypass L2 retention (keep shared weights resident).
template <bool NT>
DEV void g2s_h(const float* __restrict__ g, _Float16* s, int tid) {
  for (int i = tid; i < 1024; i += 256) {
    f32x4 v = NT ? __builtin_nontemporal_load(((const f32x4*)g) + i)
                 : ((const f32x4*)g)[i];
    h16x4 h = { (_Float16)v.x, (_Float16)v.y, (_Float16)v.z, (_Float16)v.w };
    ((h16x4*)s)[i] = h;
  }
}

extern "C" __global__ __launch_bounds__(256)
void hebbian_fused_kernel(
    const float* __restrict__ value, const float* __restrict__ key,
    const float* __restrict__ modulation, const float* __restrict__ w_assoc,
    const float* __restrict__ query, const float* __restrict__ Wk,
    const float* __restrict__ bk, const float* __restrict__ Wv,
    const float* __restrict__ bv, const float* __restrict__ A,
    const float* __restrict__ Bp, const float* __restrict__ Wq,
    const float* __restrict__ bq, const float* __restrict__ Wagg,
    const float* __restrict__ bagg, const int* __restrict__ done_idx,
    float* __restrict__ out, int b_done)
{
  extern __shared__ char smem[];
  const int tid  = threadIdx.x;
  const int lane = tid & 31;
  const int wid  = tid >> 5;
  const int b    = blockIdx.x;

  _Float16* bufA = (_Float16*)(smem + OFF_A);
  _Float16* bufB = (_Float16*)(smem + OFF_B);
  _Float16* bufC = (_Float16*)(smem + OFF_C);
  _Float16* bufD = (_Float16*)(smem + OFF_D);
  _Float16* bufE = (_Float16*)(smem + OFF_E);
  _Float16* bufF = (_Float16*)(smem + OFF_F);
  _Float16* bufG = (_Float16*)(smem + OFF_G);
  _Float16* bufH = (_Float16*)(smem + OFF_H);
  _Float16* bufI = (_Float16*)(smem + OFF_I);
  _Float16* bufJ = (_Float16*)(smem + OFF_J);
  float*    a2f  = (float*)(smem + OFF_E);     // aliases bufE+bufF
  float*    pf   = (float*)(smem + OFF_I);     // aliases bufI+bufJ
  float*    w32  = (float*)(smem + OFF_W32);
  float* sQuery  = (float*)(smem + OFF_SMALL);
  float* sMod    = sQuery + 64;
  float* sBk     = sQuery + 128;
  float* sBv     = sQuery + 192;
  _Float16* sWnT   = bufG;                     // w_new^T (N-major) for read phase
  _Float16* sQ     = bufH;                     // 16x64 padded q
  float*    sVflat = (float*)(smem + OFF_H + 2048);

  const int ti0 = wid >> 2,        tj0 = wid & 3;
  const int ti1 = (wid + 8) >> 2,  tj1 = (wid + 8) & 3;

  if (b < b_done) {
    const int widx = done_idx[b];
    // ---- epoch 0: global -> LDS (f16 operands, f32 copy of w) ----
    g2s_h<true>(key   + (size_t)b * 4096, bufA, tid);   // sKey (A op), streamed
    g2s_h<true>(value + (size_t)b * 4096, bufC, tid);   // sVal (A op), streamed
    g2s_h<false>(Wk, bufB, tid);    // sWkB = Wk row-major == N-major B of k_enc
    g2s_h<false>(Wv, bufD, tid);    // sWvB
    g2s_h<false>(A,  bufG, tid);    // sA
    g2s_h<false>(Bp, bufH, tid);    // sB
    {
      const float* wsrc = w_assoc + (size_t)widx * 4096;
      for (int i = tid; i < 1024; i += 256) {
        f32x4 v = __builtin_nontemporal_load(((const f32x4*)wsrc) + i);  // streamed
        ((f32x4*)w32)[i] = v;
        h16x4 hw = { (_Float16)v.x, (_Float16)v.y, (_Float16)v.z, (_Float16)v.w };
        ((h16x4*)bufJ)[i] = hw;                   // sW  (B of a3: w[k][u], N-major)
        h16x4 ho = { (_Float16)(1.f - v.x), (_Float16)(1.f - v.y),
                     (_Float16)(1.f - v.z), (_Float16)(1.f - v.w) };
        ((h16x4*)bufI)[i] = ho;                   // s(1-w) (B of a1, N-major)
      }
    }
    if (tid < 64) {
      sQuery[tid] = query[(size_t)b * 64 + tid];
      sMod[tid]   = modulation[(size_t)b * 64 + tid];
      sBk[tid]    = bk[tid];
      sBv[tid]    = bv[tid];
    }
    __syncthreads();

    // ---- epoch 1: k_enc = key@Wk^T + bk   (-> sKencT N-major, bufE)
    //               v_enc = (val@Wv^T + bv)*mod (-> sVencT N-major, bufF)
    { v8f x0 = mmtile(bufA, bufB, ti0, tj0, lane);
      v8f x1 = mmtile(bufA, bufB, ti1, tj1, lane);
      stN<true, false>(bufE, ti0 * 16, tj0 * 16, x0, lane, sBk, nullptr);
      stN<true, false>(bufE, ti1 * 16, tj1 * 16, x1, lane, sBk, nullptr); }
    { v8f x0 = mmtile(bufC, bufD, ti0, tj0, lane);
      v8f x1 = mmtile(bufC, bufD, ti1, tj1, lane);
      stN<true, true>(bufF, ti0 * 16, tj0 * 16, x0, lane, sBv, sMod);
      stN<true, true>(bufF, ti1 * 16, tj1 * 16, x1, lane, sBv, sMod); }
    __syncthreads();

    // ---- epoch 2 (independent): corr, reg, a1, a3 ----
    { v8f x0 = mmtile(bufE, bufF, ti0, tj0, lane);   // corr = kenc^T @ venc
      v8f x1 = mmtile(bufE, bufF, ti1, tj1, lane);
      stN<false, false>(bufB, ti0 * 16, tj0 * 16, x0, lane, nullptr, nullptr);
      stN<false, false>(bufB, ti1 * 16, tj1 * 16, x1, lane, nullptr, nullptr); }
    { v8f x0 = mmtile(bufE, bufE, ti0, tj0, lane);   // reg = kenc^T @ kenc
      v8f x1 = mmtile(bufE, bufE, ti1, tj1, lane);
      stN<false, false>(bufD, ti0 * 16, tj0 * 16, x0, lane, nullptr, nullptr);
      stN<false, false>(bufD, ti1 * 16, tj1 * 16, x1, lane, nullptr, nullptr); }
    { v8f x0 = mmtile(bufG, bufI, ti0, tj0, lane);   // a1 = A @ (1-w)^T
      v8f x1 = mmtile(bufG, bufI, ti1, tj1, lane);
      stRh(bufA, ti0 * 16, tj0 * 16, x0, lane);      // sA1 row-major
      stRh(bufA, ti1 * 16, tj1 * 16, x1, lane); }
    { v8f x0 = mmtile(bufH, bufJ, ti0, tj0, lane);   // a3 = B @ w^T
      v8f x1 = mmtile(bufH, bufJ, ti1, tj1, lane);
      stRh(bufC, ti0 * 16, tj0 * 16, x0, lane);      // sA3 row-major
      stRh(bufC, ti1 * 16, tj1 * 16, x1, lane); }
    __syncthreads();

    // ---- epoch 3: a2 = a1@corr (f32), P = a3@reg (f32; a4 = P^T) ----
    { v8f x0 = mmtile(bufA, bufB, ti0, tj0, lane);
      v8f x1 = mmtile(bufA, bufB, ti1, tj1, lane);
      stRf(a2f, ti0 * 16, tj0 * 16, x0, lane);
      stRf(a2f, ti1 * 16, tj1 * 16, x1, lane); }
    { v8f x0 = mmtile(bufC, bufD, ti0, tj0, lane);
      v8f x1 = mmtile(bufC, bufD, ti1, tj1, lane);
      stRf(pf, ti0 * 16, tj0 * 16, x0, lane);
      stRf(pf, ti1 * 16, tj1 * 16, x1, lane); }
    __syncthreads();

    // ---- weight update: w_new = w + 0.5*(a2 - P^T); store transposed f16 ----
    for (int e = tid; e < 4096; e += 256) {
      const int k = e >> 6, v = e & 63;
      const float wn = w32[e] + 0.5f * (a2f[e] - pf[v * 64 + k]);
      sWnT[v * 64 + k] = (_Float16)wn;
    }
  } else {
    // not-done batch: w_new = w_assoc[b]; stage transposed f16 for read phase
    if (tid < 64) sQuery[tid] = query[(size_t)b * 64 + tid];
    const float* wsrc = w_assoc + (size_t)b * 4096;
    for (int e = tid; e < 4096; e += 256)
      sWnT[(e & 63) * 64 + (e >> 6)] =
          (_Float16)__builtin_nontemporal_load(wsrc + e);
  }
  __syncthreads();

  // ---- read phase ----
  for (int i = tid; i < 1024; i += 256) sQ[i] = (_Float16)0.f;  // zero 16x64 pad
  __syncthreads();
  {
    float acc = bq[tid];                         // q_full[i] = Wq[i,:]·query + bq[i]
    const float* wq = Wq + (size_t)tid * 64;
#pragma unroll 8
    for (int c = 0; c < 64; ++c) acc += sQuery[c] * wq[c];
    sQ[(tid >> 6) * 64 + (tid & 63)] = (_Float16)acc;   // [h][k], rows 4..15 zero
  }
  __syncthreads();
  if (wid < 4) {                                 // v = q @ w_new via WMMA (full waves)
    v8f acc = mmtile(sQ, sWnT, 0, wid, lane);
    if (lane < 16) {
#pragma unroll
      for (int i = 0; i < 4; ++i)
        sVflat[i * 64 + wid * 16 + lane] = acc[i];      // v_flat[h*64+vv]
    }
  }
  __syncthreads();
  if (tid < 64) {                                // out = Wagg @ v_flat + bagg
    float acc = bagg[tid];
    const float* wa = Wagg + (size_t)tid * 256;
#pragma unroll 8
    for (int i = 0; i < 256; ++i) acc += wa[i] * sVflat[i];
    __builtin_nontemporal_store(acc, out + (size_t)b * 64 + tid);
  }
}

extern "C" void kernel_launch(void* const* d_in, const int* in_sizes, int n_in,
                              void* d_out, int out_size, void* d_ws, size_t ws_size,
                              hipStream_t stream) {
  const float* value      = (const float*)d_in[0];
  const float* key        = (const float*)d_in[1];
  const float* modulation = (const float*)d_in[2];
  const float* w_assoc    = (const float*)d_in[3];
  const float* query      = (const float*)d_in[4];
  const float* Wk         = (const float*)d_in[5];
  const float* bk         = (const float*)d_in[6];
  const float* Wv         = (const float*)d_in[7];
  const float* bv         = (const float*)d_in[8];
  const float* A          = (const float*)d_in[9];
  const float* Bp         = (const float*)d_in[10];
  const float* Wq         = (const float*)d_in[11];
  const float* bq         = (const float*)d_in[12];
  const float* Wagg       = (const float*)d_in[13];
  const float* bagg       = (const float*)d_in[14];
  const int*   done_idx   = (const int*)d_in[15];
  float* out = (float*)d_out;

  const int b_done = in_sizes[0] / (64 * 64);   // value:  [B_DONE, 64, 64]
  const int b_full = in_sizes[4] / 64;          // query:  [B_FULL, 64]

  // Idempotent, deterministic, host-side; required for >64KB dynamic LDS.
  (void)hipFuncSetAttribute((const void*)hebbian_fused_kernel,
                            hipFuncAttributeMaxDynamicSharedMemorySize, SMEM_BYTES);

  hebbian_fused_kernel<<<dim3(b_full), dim3(256), SMEM_BYTES, stream>>>(
      value, key, modulation, w_assoc, query, Wk, bk, Wv, bv, A, Bp,
      Wq, bq, Wagg, bagg, done_idx, out, b_done);
}
